// LSTMReset_17300128268689
// MI455X (gfx1250) — matrix operand order repaired and checked by reading, not compiled, not run on hardware
//
#include <hip/hip_runtime.h>
#include <cstdint>

#define T_STEPS 256
#define B_SZ    64
#define D_IN    512
#define H_SZ    1024
#define G4H     4096   /* 4*H */

typedef __attribute__((ext_vector_type(16))) __bf16   v16bf;
typedef __attribute__((ext_vector_type(8)))  float    v8f;
typedef __attribute__((ext_vector_type(4)))  unsigned u32x4;

union FragU { v16bf v; u32x4 q[2]; };

__device__ __forceinline__ float sigmoidf_(float x) {
    return 1.0f / (1.0f + __expf(-x));
}
__device__ __forceinline__ float tanhf_(float x) {
    // tanh(x) = 1 - 2/(exp(2x)+1)
    return 1.0f - 2.0f / (__expf(2.0f * x) + 1.0f);
}

// ---------------- setup kernels ----------------

__global__ void k_f32_to_bf16(const float* __restrict__ src, __bf16* __restrict__ dst, int n) {
    int i = blockIdx.x * blockDim.x + threadIdx.x;
    int stride = gridDim.x * blockDim.x;
    for (; i < n; i += stride) dst[i] = (__bf16)src[i];
}

// dst[j, 0:kx] = Wih[j,:], dst[j, kx:kx+kh] = Whh[j,:]  (bf16, row-major, K = kx+kh)
__global__ void k_build_wc(const float* __restrict__ wih, const float* __restrict__ whh,
                           __bf16* __restrict__ dst, int kx, int kh) {
    const int K = kx + kh;
    const long long n = (long long)G4H * K;
    long long i = (long long)blockIdx.x * blockDim.x + threadIdx.x;
    const long long stride = (long long)gridDim.x * blockDim.x;
    for (; i < n; i += stride) {
        int j = (int)(i / K);
        int k = (int)(i % K);
        float v = (k < kx) ? wih[(long long)j * kx + k]
                           : whh[(long long)j * kh + (k - kx)];
        dst[i] = (__bf16)v;
    }
}

__global__ void k_bias_sum(const float* __restrict__ a, const float* __restrict__ b,
                           float* __restrict__ dst, int n) {
    int i = blockIdx.x * blockDim.x + threadIdx.x;
    int stride = gridDim.x * blockDim.x;
    for (; i < n; i += stride) dst[i] = a[i] + b[i];
}

// h0/c0 are [B, L=2, H] fp32; split into per-layer state buffers
__global__ void k_init_state(const float* __restrict__ h0, const float* __restrict__ c0,
                             __bf16* __restrict__ h0b, __bf16* __restrict__ h1b,
                             float* __restrict__ c0b, float* __restrict__ c1b) {
    int i = blockIdx.x * blockDim.x + threadIdx.x;
    int stride = gridDim.x * blockDim.x;
    const int n = B_SZ * H_SZ;
    for (; i < n; i += stride) {
        int b = i >> 10;          // /H
        int h = i & (H_SZ - 1);
        long long s = (long long)b * (2 * H_SZ) + h;
        h0b[i] = (__bf16)h0[s];
        h1b[i] = (__bf16)h0[s + H_SZ];
        c0b[i] = c0[s];
        c1b[i] = c0[s + H_SZ];
    }
}

// ---------------- fused LSTM step (one layer, one timestep) ----------------

__device__ __forceinline__ void load_a(FragU& a, const __bf16* __restrict__ row,
                                       int kA, int laneHi, unsigned msk) {
    const int ka = kA + laneHi * 8;
    a.q[0] = *(const u32x4*)(row + ka)      & (u32x4)msk;
    a.q[1] = *(const u32x4*)(row + ka + 16) & (u32x4)msk;
}
__device__ __forceinline__ void load_b(FragU& b, const __bf16* __restrict__ w,
                                       int kW, int laneHi) {
    const int kw = kW + laneHi * 8;
    b.q[0] = *(const u32x4*)(w + kw);
    b.q[1] = *(const u32x4*)(w + kw + 16);
}

#define WMMA_BF16(ACC, A, Bf) \
    ACC = __builtin_amdgcn_wmma_f32_16x16x32_bf16(false, (A).v, false, (Bf).v, (short)0, ACC, false, false)

// One GEMM phase over kcount (multiple of 64) K-columns.
// 2 M-subtiles (rows row0/row1) share the 4 gate B-fragments; 2-stage software pipeline.
__device__ __forceinline__ void gemm_phase(const __bf16* __restrict__ row0,
                                           const __bf16* __restrict__ row1,
                                           const __bf16* __restrict__ w0,
                                           const __bf16* __restrict__ w1,
                                           const __bf16* __restrict__ w2,
                                           const __bf16* __restrict__ w3,
                                           int kbase, int kcount, int laneHi,
                                           unsigned msk0, unsigned msk1,
                                           v8f& a00, v8f& a01, v8f& a02, v8f& a03,
                                           v8f& a10, v8f& a11, v8f& a12, v8f& a13) {
    FragU xA0, xA1, bA0, bA1, bA2, bA3;   // stage A
    FragU xB0, xB1, bB0, bB1, bB2, bB3;   // stage B
    load_a(xA0, row0, 0, laneHi, msk0);
    load_a(xA1, row1, 0, laneHi, msk1);
    load_b(bA0, w0, kbase, laneHi); load_b(bA1, w1, kbase, laneHi);
    load_b(bA2, w2, kbase, laneHi); load_b(bA3, w3, kbase, laneHi);
    for (int k = 0; k < kcount; k += 64) {
        // issue stage-B loads before consuming stage A
        load_a(xB0, row0, k + 32, laneHi, msk0);
        load_a(xB1, row1, k + 32, laneHi, msk1);
        load_b(bB0, w0, kbase + k + 32, laneHi); load_b(bB1, w1, kbase + k + 32, laneHi);
        load_b(bB2, w2, kbase + k + 32, laneHi); load_b(bB3, w3, kbase + k + 32, laneHi);
        WMMA_BF16(a00, xA0, bA0); WMMA_BF16(a10, xA1, bA0);
        WMMA_BF16(a01, xA0, bA1); WMMA_BF16(a11, xA1, bA1);
        WMMA_BF16(a02, xA0, bA2); WMMA_BF16(a12, xA1, bA2);
        WMMA_BF16(a03, xA0, bA3); WMMA_BF16(a13, xA1, bA3);
        if (k + 64 < kcount) {    // uniform scalar branch
            load_a(xA0, row0, k + 64, laneHi, msk0);
            load_a(xA1, row1, k + 64, laneHi, msk1);
            load_b(bA0, w0, kbase + k + 64, laneHi); load_b(bA1, w1, kbase + k + 64, laneHi);
            load_b(bA2, w2, kbase + k + 64, laneHi); load_b(bA3, w3, kbase + k + 64, laneHi);
        }
        WMMA_BF16(a00, xB0, bB0); WMMA_BF16(a10, xB1, bB0);
        WMMA_BF16(a01, xB0, bB1); WMMA_BF16(a11, xB1, bB1);
        WMMA_BF16(a02, xB0, bB2); WMMA_BF16(a12, xB1, bB2);
        WMMA_BF16(a03, xB0, bB3); WMMA_BF16(a13, xB1, bB3);
    }
}

// Fused gate nonlinearity + state update for one 16x16 (M-subtile x N) tile.
__device__ __forceinline__ void epilogue_sub(const v8f& gi, const v8f& gf, const v8f& gg, const v8f& go,
                                             int msub, int laneHi, int ncol,
                                             float bi, float bf, float bg, float bo,
                                             const int* __restrict__ reset_t,
                                             float* __restrict__ cbuf,
                                             __bf16* __restrict__ hout,
                                             float* __restrict__ out_f32,
                                             float* __restrict__ hN,
                                             float* __restrict__ cN) {
#pragma unroll
    for (int r = 0; r < 8; ++r) {
        const int m = msub + laneHi * 8 + r;
        const float keep = reset_t[m] ? 0.0f : 1.0f;
        const long long idx = (long long)m * H_SZ + ncol;
        const float cprev = cbuf[idx] * keep;
        const float ig = sigmoidf_(gi[r] + bi);
        const float fg = sigmoidf_(gf[r] + bf);
        const float g_ = tanhf_(gg[r] + bg);
        const float og = sigmoidf_(go[r] + bo);
        const float cn = fg * cprev + ig * g_;
        const float hn = og * tanhf_(cn);
        cbuf[idx] = cn;
        hout[idx] = (__bf16)hn;
        if (out_f32) out_f32[idx] = hn;
        const long long idx2 = (long long)m * (2 * H_SZ) + ncol;
        if (hN) hN[idx2] = hn;
        if (cN) cN[idx2] = cn;
    }
}

// gates[b, g*H+n] = sum_k A[b,k] * Wc[g*H+n, k] + bias[g*H+n]
//   A = [ x-region (k < split, from xpart) | h-region (k >= split, from hprev, reset-masked) ]
// Grid: 128 blocks x 32 threads; 1 wave = one 32(batch) x 16(hidden) tile,
// 2 M-subtiles share B fragments -> 8 WMMA accumulators, half the weight traffic.
__global__ __launch_bounds__(32)
void k_lstm_step(const __bf16* __restrict__ Wc,      // [4H, K] bf16 row-major
                 const float*  __restrict__ bias,    // [4H]
                 const __bf16* __restrict__ xpart,   // [B, xstride]
                 int xstride,
                 const __bf16* __restrict__ hprev,   // [B, H]
                 float*        __restrict__ cbuf,    // [B, H] in/out
                 __bf16*       __restrict__ hout,    // [B, H] next-step h (bf16)
                 const int*    __restrict__ reset_t, // [B]
                 int K, int split,
                 float* __restrict__ out_f32,        // optional [B, H]
                 float* __restrict__ hN,             // optional, row stride 2H
                 float* __restrict__ cN)             // optional, row stride 2H
{
    const int lane    = threadIdx.x & 31;
    const int laneLow = lane & 15;
    const int laneHi  = lane >> 4;
    const int tile    = blockIdx.x;
    const int m0      = (tile & 1) * 32;   // batch tile (32 rows)
    const int n0      = (tile >> 1) * 16;  // hidden-column tile

    const int arow0 = m0 + laneLow;        // A rows this lane supplies (2 subtiles)
    const int arow1 = arow0 + 16;
    const unsigned msk0 = reset_t[arow0] ? 0u : ~0u;
    const unsigned msk1 = reset_t[arow1] ? 0u : ~0u;

    const __bf16* xr0 = xpart + (long long)arow0 * xstride;
    const __bf16* xr1 = xpart + (long long)arow1 * xstride;
    const __bf16* hr0 = hprev + (long long)arow0 * H_SZ;
    const __bf16* hr1 = hprev + (long long)arow1 * H_SZ;

    const __bf16* w0 = Wc + (long long)(0 * H_SZ + n0 + laneLow) * K;
    const __bf16* w1 = Wc + (long long)(1 * H_SZ + n0 + laneLow) * K;
    const __bf16* w2 = Wc + (long long)(2 * H_SZ + n0 + laneLow) * K;
    const __bf16* w3 = Wc + (long long)(3 * H_SZ + n0 + laneLow) * K;

    v8f a00 = (v8f)0.0f, a01 = (v8f)0.0f, a02 = (v8f)0.0f, a03 = (v8f)0.0f;  // subtile 0: i,f,g,o
    v8f a10 = (v8f)0.0f, a11 = (v8f)0.0f, a12 = (v8f)0.0f, a13 = (v8f)0.0f;  // subtile 1: i,f,g,o

    // phase 1: x-region, K-range [0, split)   (no reset mask)
    gemm_phase(xr0, xr1, w0, w1, w2, w3, 0, split, laneHi, ~0u, ~0u,
               a00, a01, a02, a03, a10, a11, a12, a13);
    // phase 2: h-region, K-range [split, K)   (reset-masked rows)
    gemm_phase(hr0, hr1, w0, w1, w2, w3, split, K - split, laneHi, msk0, msk1,
               a00, a01, a02, a03, a10, a11, a12, a13);

    const int ncol = n0 + laneLow;
    const float bi = bias[ncol];
    const float bf = bias[H_SZ + ncol];
    const float bg = bias[2 * H_SZ + ncol];
    const float bo = bias[3 * H_SZ + ncol];

    epilogue_sub(a00, a01, a02, a03, m0,      laneHi, ncol, bi, bf, bg, bo,
                 reset_t, cbuf, hout, out_f32, hN, cN);
    epilogue_sub(a10, a11, a12, a13, m0 + 16, laneHi, ncol, bi, bf, bg, bo,
                 reset_t, cbuf, hout, out_f32, hN, cN);
}

// ---------------- host ----------------

extern "C" void kernel_launch(void* const* d_in, const int* in_sizes, int n_in,
                              void* d_out, int out_size, void* d_ws, size_t ws_size,
                              hipStream_t stream) {
    (void)in_sizes; (void)n_in; (void)out_size; (void)ws_size;

    const float* x     = (const float*)d_in[0];   // [T,B,D]
    const int*   reset = (const int*)  d_in[1];   // [T,B]
    const float* h0    = (const float*)d_in[2];   // [B,2,H]
    const float* c0    = (const float*)d_in[3];   // [B,2,H]
    const float* Wih0  = (const float*)d_in[4];
    const float* Whh0  = (const float*)d_in[5];
    const float* bih0  = (const float*)d_in[6];
    const float* bhh0  = (const float*)d_in[7];
    const float* Wih1  = (const float*)d_in[8];
    const float* Whh1  = (const float*)d_in[9];
    const float* bih1  = (const float*)d_in[10];
    const float* bhh1  = (const float*)d_in[11];

    char* ws = (char*)d_ws;
    size_t off = 0;
    auto alloc = [&](size_t bytes) { size_t r = off; off = (off + bytes + 255) & ~(size_t)255; return r; };

    const size_t oWc0  = alloc((size_t)G4H * 1536 * 2);
    const size_t oWc1  = alloc((size_t)G4H * 2048 * 2);
    const size_t oXb   = alloc((size_t)T_STEPS * B_SZ * D_IN * 2);
    const size_t oB0   = alloc((size_t)G4H * 4);
    const size_t oB1   = alloc((size_t)G4H * 4);
    const size_t oH0a  = alloc((size_t)B_SZ * H_SZ * 2);
    const size_t oH0b  = alloc((size_t)B_SZ * H_SZ * 2);
    const size_t oH1a  = alloc((size_t)B_SZ * H_SZ * 2);
    const size_t oH1b  = alloc((size_t)B_SZ * H_SZ * 2);
    const size_t oC0   = alloc((size_t)B_SZ * H_SZ * 4);
    const size_t oC1   = alloc((size_t)B_SZ * H_SZ * 4);

    __bf16* Wc0   = (__bf16*)(ws + oWc0);
    __bf16* Wc1   = (__bf16*)(ws + oWc1);
    __bf16* Xb    = (__bf16*)(ws + oXb);
    float*  bias0 = (float*)(ws + oB0);
    float*  bias1 = (float*)(ws + oB1);
    __bf16* h0buf[2] = { (__bf16*)(ws + oH0a), (__bf16*)(ws + oH0b) };
    __bf16* h1buf[2] = { (__bf16*)(ws + oH1a), (__bf16*)(ws + oH1b) };
    float*  c0buf = (float*)(ws + oC0);
    float*  c1buf = (float*)(ws + oC1);

    // ---- setup (recomputed every call; deterministic) ----
    k_build_wc<<<4096, 256, 0, stream>>>(Wih0, Whh0, Wc0, D_IN, H_SZ);
    k_build_wc<<<4096, 256, 0, stream>>>(Wih1, Whh1, Wc1, H_SZ, H_SZ);
    k_f32_to_bf16<<<8192, 256, 0, stream>>>(x, Xb, T_STEPS * B_SZ * D_IN);
    k_bias_sum<<<16, 256, 0, stream>>>(bih0, bhh0, bias0, G4H);
    k_bias_sum<<<16, 256, 0, stream>>>(bih1, bhh1, bias1, G4H);
    k_init_state<<<256, 256, 0, stream>>>(h0, c0, h0buf[0], h1buf[0], c0buf, c1buf);

    float* outp = (float*)d_out;                               // [T,B,H]
    float* hN   = outp + (size_t)T_STEPS * B_SZ * H_SZ;        // [B,2,H]
    float* cN   = hN + (size_t)B_SZ * 2 * H_SZ;                // [B,2,H]

    // ---- recurrence: 2 layer-step kernels per timestep, stream-ordered ----
    for (int t = 0; t < T_STEPS; ++t) {
        const int rp = t & 1;
        const int wp = 1 - rp;
        const int*  rt   = reset + (size_t)t * B_SZ;
        const bool  last = (t == T_STEPS - 1);

        // layer 0: A = [ x_t | h0_prev*keep ], K = 512 + 1024
        k_lstm_step<<<128, 32, 0, stream>>>(
            Wc0, bias0,
            Xb + (size_t)t * B_SZ * D_IN, D_IN,
            h0buf[rp], c0buf, h0buf[wp], rt,
            1536, 512,
            nullptr,
            last ? (hN + 0)    : nullptr,
            last ? (cN + 0)    : nullptr);

        // layer 1: A = [ h0_new | h1_prev*keep ], K = 1024 + 1024
        k_lstm_step<<<128, 32, 0, stream>>>(
            Wc1, bias1,
            h0buf[wp], H_SZ,
            h1buf[rp], c1buf, h1buf[wp], rt,
            2048, 1024,
            outp + (size_t)t * B_SZ * H_SZ,
            last ? (hN + H_SZ) : nullptr,
            last ? (cN + H_SZ) : nullptr);
    }
}
